// SelfAttentionBlock_16604343567002
// MI455X (gfx1250) — compile-verified, matrix operand order checked by reading, not run on hardware
//
#include <hip/hip_runtime.h>
#include <cstdint>
#include <cstddef>

typedef __attribute__((ext_vector_type(16))) _Float16 v16h;
typedef __attribute__((ext_vector_type(8)))  _Float16 v8h;
typedef __attribute__((ext_vector_type(8)))  float    v8f;
typedef __attribute__((ext_vector_type(4)))  unsigned int v4u;
typedef __attribute__((ext_vector_type(8)))  int      v8i;
typedef __attribute__((ext_vector_type(4)))  int      v4i;

#define DIM    256
#define HEADS  8
#define DHEAD  32
#define SEQ    2048
#define BATCH  2
#define ROWS   (BATCH * SEQ)   // 4096
#define FF     512
#define LN_EPS 1e-5f

// ---------------------------------------------------------------------------
// Fragment loader (global or LDS): A operand (16x32, M x K) and "Bt" operand.
// CDNA5 16-bit A layout: lane L (row = L%16, hlf = L/16) holds
//   K = hlf*8 + 0..7 and K = 16 + hlf*8 + 0..7  => two 16-byte loads per lane.
// ---------------------------------------------------------------------------
__device__ __forceinline__ v16h load_frag(const _Float16* p, int ld, int row0, int k0) {
  const int lane = threadIdx.x & 31;
  const int r    = row0 + (lane & 15);
  const int hlf  = lane >> 4;
  const _Float16* base = p + (size_t)r * ld + k0 + hlf * 8;
  v8h lo = *(const v8h*)(base);
  v8h hi = *(const v8h*)(base + 16);
  v16h out;
#pragma unroll
  for (int i = 0; i < 8; ++i) { out[i] = lo[i]; out[8 + i] = hi[i]; }
  return out;
}

// ---------------------------------------------------------------------------
// Tensor Data Mover: async 2D tile (f16 elements) global -> LDS.
// Builds the D# per CDNA5 ISA (group0: count/lds_addr/global_addr/type=2,
// group1: data_size=2B, tensor dims, tile dims, dim0 stride). Tracked by
// TENSORcnt; caller must s_wait_tensorcnt and barrier before consuming.
// ---------------------------------------------------------------------------
__device__ __forceinline__ void tdm_load_2d_f16(const _Float16* gptr, unsigned lds_off,
                                                unsigned width, unsigned height,
                                                unsigned stride) {
  const unsigned long long ga = (unsigned long long)(size_t)gptr;
  v4u g0;
  g0[0] = 1u;                                              // count=1, user D#
  g0[1] = lds_off;                                         // lds_addr (bytes)
  g0[2] = (unsigned)(ga & 0xffffffffu);                    // global_addr[31:0]
  g0[3] = (unsigned)((ga >> 32) & 0x1ffffffu) | (2u << 30);// [56:32] + type=2
  v8i g1;
  g1[0] = (int)(1u << 16);                                 // data_size=1 (2B)
  g1[1] = (int)((width & 0xffffu) << 16);                  // tensor_dim0[15:0]
  g1[2] = (int)(((width >> 16) & 0xffffu) | ((height & 0xffffu) << 16));
  g1[3] = (int)(((height >> 16) & 0xffffu) | ((width & 0xffffu) << 16)); // tile_dim0
  g1[4] = (int)(height & 0xffffu);                         // tile_dim1 (tile_dim2=0)
  g1[5] = (int)(stride);                                   // dim0_stride[31:0]
  g1[6] = 0;                                               // stride[47:32]=0
  g1[7] = 0;
  v4i g2 = {};
  v4i g3 = {};
#if __clang_major__ >= 23
  v8i g4 = {};
  __builtin_amdgcn_tensor_load_to_lds(g0, g1, g2, g3, g4, 0);
#else
  __builtin_amdgcn_tensor_load_to_lds(g0, g1, g2, g3, 0);
#endif
}

// ---------------------------------------------------------------------------
// Generic WMMA GEMM: C[M x N] = A[M x K] * Bt[N x K]^T, f16 in, f32 accum.
// 8 waves per block, one 16x16 output tile per wave.
// ---------------------------------------------------------------------------
constexpr int EPI_F32       = 0;
constexpr int EPI_F16       = 1;
constexpr int EPI_RELU_F16  = 2;
constexpr int EPI_RESID_F32 = 3;

template <int EPI>
__global__ void __launch_bounds__(256)
gemm_wmma(const _Float16* __restrict__ A, const _Float16* __restrict__ Bt,
          void* __restrict__ Cout, const float* __restrict__ resid,
          int M, int N, int K) {
  const int tiles_n = N >> 4;
  const int total   = (M >> 4) * tiles_n;
  const int tile    = blockIdx.x * 8 + (threadIdx.x >> 5);
  if (tile >= total) return;                 // wave-uniform: EXEC stays all-1s
  const int mt   = tile / tiles_n;
  const int nt   = tile - mt * tiles_n;
  const int lane = threadIdx.x & 31;

  v8f acc = {};
  for (int k0 = 0; k0 < K; k0 += 32) {
    v16h a = load_frag(A,  K, mt << 4, k0);
    v16h b = load_frag(Bt, K, nt << 4, k0);
    if (k0 + 32 < K) {
      __builtin_prefetch(A  + (size_t)((mt << 4) + (lane & 15)) * K + k0 + 32, 0, 1);
      __builtin_prefetch(Bt + (size_t)((nt << 4) + (lane & 15)) * K + k0 + 32, 0, 1);
    }
    acc = __builtin_amdgcn_wmma_f32_16x16x32_f16(false, a, false, b,
                                                 (short)0, acc, false, false);
  }

  const int col = (nt << 4) + (lane & 15);
  const int hlf = lane >> 4;
#pragma unroll
  for (int r = 0; r < 8; ++r) {
    const int row = (mt << 4) + hlf * 8 + r;   // C layout: VGPR r -> M = r + 8*hlf
    float v = acc[r];
    if (EPI == EPI_RESID_F32) v += resid[(size_t)row * N + col];
    if (EPI == EPI_RELU_F16)  v = v > 0.f ? v : 0.f;
    if (EPI == EPI_F32 || EPI == EPI_RESID_F32)
      ((float*)Cout)[(size_t)row * N + col] = v;
    else
      ((_Float16*)Cout)[(size_t)row * N + col] = (_Float16)v;
  }
}

// ---------------------------------------------------------------------------
// Workgroup-cooperative flash attention: 8 waves handle 128 query rows of one
// (batch, head). Per 32-key step the TDM stages K (32x32) and Vt (256x32)
// tiles into LDS (shared by all 8 waves -> 8x L2-traffic reduction); each wave
// runs 2 WMMAs for S = Q*K^T, online softmax (shfl_xor reductions), transposes
// P through its private LDS slab into A-fragment layout, then 16 WMMAs for
// O += P*V. All WMMAs execute with full EXEC (uniform control flow only).
// ---------------------------------------------------------------------------
__global__ void __launch_bounds__(256)
attn_wmma(const _Float16* __restrict__ Q, const _Float16* __restrict__ Km,
          const _Float16* __restrict__ Vt, _Float16* __restrict__ Out) {
  const int qb = blockIdx.x, h = blockIdx.y, b = blockIdx.z;
  const int wave = threadIdx.x >> 5;
  const int lane = threadIdx.x & 31;
  const int hlf  = lane >> 4;
  const float scale = 0.17677669529663687f;   // 32^-0.5
  const float LOG2E = 1.4426950408889634f;

  const _Float16* Qb  = Q  + (size_t)b * SEQ * DIM;
  const _Float16* Kb  = Km + (size_t)b * SEQ * DIM;
  const _Float16* Vtb = Vt + (size_t)b * DIM * SEQ;

  __shared__ _Float16 sK[32 * 32];         // K tile: 32 keys x 32 dims (2 KB)
  __shared__ _Float16 sV[DIM * 32];        // Vt tile: 256 rows x 32 keys (16 KB)
  __shared__ _Float16 sP[8][16 * 32];      // per-wave P transpose slab (8 KB)

  const int qrow0 = (qb << 7) + (wave << 4);        // this wave's 16 query rows
  const v16h qf = load_frag(Qb, DIM, qrow0, h * DHEAD);

  v8f zero = {};
  v8f o[16];
#pragma unroll
  for (int t = 0; t < 16; ++t) o[t] = zero;
  float mrow[8], lrow[8];
#pragma unroll
  for (int r = 0; r < 8; ++r) { mrow[r] = -3.0e38f; lrow[r] = 0.f; }

  for (int j0 = 0; j0 < SEQ; j0 += 32) {
    __syncthreads();                       // LDS consumers of prev step done
    if (wave == 0) {                       // one wave drives the TDM
      tdm_load_2d_f16(Kb + (size_t)j0 * DIM + h * DHEAD,
                      (unsigned)(size_t)&sK[0], 32, 32, DIM);
      tdm_load_2d_f16(Vtb + j0,
                      (unsigned)(size_t)&sV[0], 32, DIM, SEQ);
      __builtin_amdgcn_s_wait_tensorcnt(0);
    }
    __syncthreads();                       // tiles visible to all waves

    v16h kf0 = load_frag(&sK[0], 32, 0,  0);
    v16h kf1 = load_frag(&sK[0], 32, 16, 0);
    v8f s0 = __builtin_amdgcn_wmma_f32_16x16x32_f16(false, qf, false, kf0,
                                                    (short)0, zero, false, false);
    v8f s1 = __builtin_amdgcn_wmma_f32_16x16x32_f16(false, qf, false, kf1,
                                                    (short)0, zero, false, false);
#pragma unroll
    for (int r = 0; r < 8; ++r) {
      const float x0 = s0[r] * scale;
      const float x1 = s1[r] * scale;
      float mx = fmaxf(x0, x1);
#pragma unroll
      for (int m = 1; m < 16; m <<= 1) mx = fmaxf(mx, __shfl_xor(mx, m, 32));
      const float mnew = fmaxf(mrow[r], mx);
      const float corr = exp2f((mrow[r] - mnew) * LOG2E);
      const float p0   = exp2f((x0 - mnew) * LOG2E);
      const float p1   = exp2f((x1 - mnew) * LOG2E);
      float ps = p0 + p1;
#pragma unroll
      for (int m = 1; m < 16; m <<= 1) ps += __shfl_xor(ps, m, 32);
      lrow[r] = lrow[r] * corr + ps;
      mrow[r] = mnew;
#pragma unroll
      for (int t = 0; t < 16; ++t) o[t][r] *= corr;
      // write P row-major 16x32 into this wave's slab (cross-lane transpose)
      sP[wave][(hlf * 8 + r) * 32 + (lane & 15)]      = (_Float16)p0;
      sP[wave][(hlf * 8 + r) * 32 + 16 + (lane & 15)] = (_Float16)p1;
    }
    const v16h pf = load_frag(&sP[wave][0], 32, 0, 0);   // ds_load_b128 x2
#pragma unroll
    for (int t = 0; t < 16; ++t) {
      v16h vf = load_frag(&sV[0], 32, t << 4, 0);
      o[t] = __builtin_amdgcn_wmma_f32_16x16x32_f16(false, pf, false, vf,
                                                    (short)0, o[t], false, false);
    }
  }

  // epilogue: O /= l, store as (b, n, h*256 + col) f16
#pragma unroll
  for (int r = 0; r < 8; ++r) {
    const float inv = 1.f / lrow[r];
    const int row = qrow0 + hlf * 8 + r;
#pragma unroll
    for (int t = 0; t < 16; ++t) {
      const int col = h * DIM + (t << 4) + (lane & 15);
      Out[((size_t)(b * SEQ + row)) * (HEADS * DIM) + col] = (_Float16)(o[t][r] * inv);
    }
  }
}

// ---------------------------------------------------------------------------
// LayerNorm over last dim (256): one block per row, shuffle + LDS reduce.
// ---------------------------------------------------------------------------
__global__ void __launch_bounds__(256)
layernorm_k(const float* __restrict__ X, const float* __restrict__ gamma,
            float* __restrict__ outF32, _Float16* __restrict__ outF16) {
  const int row = blockIdx.x;
  const int tid = threadIdx.x;
  const float v = X[(size_t)row * DIM + tid];
  float s = v, s2 = v * v;
#pragma unroll
  for (int m = 1; m < 32; m <<= 1) {
    s  += __shfl_xor(s,  m, 32);
    s2 += __shfl_xor(s2, m, 32);
  }
  __shared__ float w1[8], w2[8];
  if ((tid & 31) == 0) { w1[tid >> 5] = s; w2[tid >> 5] = s2; }
  __syncthreads();
  float ts = 0.f, ts2 = 0.f;
#pragma unroll
  for (int i = 0; i < 8; ++i) { ts += w1[i]; ts2 += w2[i]; }
  const float mu  = ts  * (1.f / DIM);
  const float var = ts2 * (1.f / DIM) - mu * mu;
  const float y   = (v - mu) * rsqrtf(var + LN_EPS) * gamma[tid];
  if (outF32) outF32[(size_t)row * DIM + tid] = y;
  if (outF16) outF16[(size_t)row * DIM + tid] = (_Float16)y;
}

// ---------------------------------------------------------------------------
// Helpers: f32->f16 convert, convert+transpose (K x N -> N x K), f16 transpose
// ---------------------------------------------------------------------------
__global__ void __launch_bounds__(256)
cvt_f2h(const float* __restrict__ s, _Float16* __restrict__ d, int n) {
  const int i = blockIdx.x * 256 + threadIdx.x;
  if (i < n) d[i] = (_Float16)s[i];
}

__global__ void __launch_bounds__(256)
cvtT_f2h(const float* __restrict__ W, _Float16* __restrict__ Wt, int K, int N) {
  const int i = blockIdx.x * 256 + threadIdx.x;
  if (i < K * N) {
    const int k = i / N, nn = i - k * N;
    Wt[(size_t)nn * K + k] = (_Float16)W[i];
  }
}

__global__ void __launch_bounds__(256)
transpose_h(const _Float16* __restrict__ src, _Float16* __restrict__ dst,
            int rows, int cols) {
  __shared__ _Float16 tile[32][33];
  const int b = blockIdx.z;
  src += (size_t)b * rows * cols;
  dst += (size_t)b * rows * cols;
  const int c0 = blockIdx.x * 32;
  const int r0 = blockIdx.y * 32;
  const int tx = threadIdx.x, ty = threadIdx.y;
  for (int j = ty; j < 32; j += 8)
    tile[j][tx] = src[(size_t)(r0 + j) * cols + c0 + tx];
  __syncthreads();
  for (int j = ty; j < 32; j += 8)
    dst[(size_t)(c0 + j) * rows + r0 + tx] = tile[tx][j];
}

// ---------------------------------------------------------------------------
extern "C" void kernel_launch(void* const* d_in, const int* in_sizes, int n_in,
                              void* d_out, int out_size, void* d_ws, size_t ws_size,
                              hipStream_t stream) {
  (void)in_sizes; (void)n_in; (void)out_size; (void)ws_size;
  const float* x      = (const float*)d_in[0];
  const float* Wq     = (const float*)d_in[1];
  const float* Wk     = (const float*)d_in[2];
  const float* Wv     = (const float*)d_in[3];
  const float* Wo     = (const float*)d_in[4];
  const float* Wf1    = (const float*)d_in[5];
  const float* Wf2    = (const float*)d_in[6];
  const float* gamma1 = (const float*)d_in[7];
  const float* gamma2 = (const float*)d_in[8];

  char*  ws  = (char*)d_ws;
  size_t off = 0;
  auto take = [&](size_t bytes) -> char* {
    char* p = ws + off;
    off += (bytes + 255) & ~(size_t)255;
    return p;
  };
  _Float16* Xh    = (_Float16*)take((size_t)ROWS * DIM * 2);
  _Float16* Wq_t  = (_Float16*)take((size_t)DIM * DIM * 2);
  _Float16* Wk_t  = (_Float16*)take((size_t)DIM * DIM * 2);
  _Float16* Wv_t  = (_Float16*)take((size_t)DIM * DIM * 2);
  _Float16* Wo_t  = (_Float16*)take((size_t)DIM * HEADS * DIM * 2);   // 256 x 2048
  _Float16* Wf1_t = (_Float16*)take((size_t)FF * DIM * 2);            // 512 x 256
  _Float16* Wf2_t = (_Float16*)take((size_t)DIM * FF * 2);            // 256 x 512
  _Float16* Qh    = (_Float16*)take((size_t)ROWS * DIM * 2);
  _Float16* Kh    = (_Float16*)take((size_t)ROWS * DIM * 2);
  _Float16* Vh    = (_Float16*)take((size_t)ROWS * DIM * 2);
  _Float16* Vt    = (_Float16*)take((size_t)BATCH * DIM * SEQ * 2);
  _Float16* AttnO = (_Float16*)take((size_t)ROWS * HEADS * DIM * 2);  // 4096 x 2048
  float*    Y0    = (float*)   take((size_t)ROWS * DIM * 4);
  _Float16* Y1h   = (_Float16*)take((size_t)ROWS * DIM * 2);
  _Float16* Hb    = (_Float16*)take((size_t)ROWS * FF * 2);
  float*    Y2    = (float*)   take((size_t)ROWS * DIM * 4);

  // 1) precision conversion + weight transposes
  cvt_f2h<<<(ROWS * DIM) / 256, 256, 0, stream>>>(x, Xh, ROWS * DIM);
  cvtT_f2h<<<(DIM * DIM) / 256, 256, 0, stream>>>(Wq, Wq_t, DIM, DIM);
  cvtT_f2h<<<(DIM * DIM) / 256, 256, 0, stream>>>(Wk, Wk_t, DIM, DIM);
  cvtT_f2h<<<(DIM * DIM) / 256, 256, 0, stream>>>(Wv, Wv_t, DIM, DIM);
  cvtT_f2h<<<(DIM * HEADS * DIM) / 256, 256, 0, stream>>>(Wo, Wo_t, DIM * HEADS, DIM);
  cvtT_f2h<<<(DIM * FF) / 256, 256, 0, stream>>>(Wf1, Wf1_t, DIM, FF);
  cvtT_f2h<<<(FF * DIM) / 256, 256, 0, stream>>>(Wf2, Wf2_t, FF, DIM);

  // 2) Q/K/V projections (WMMA, f16 out)
  const int tilesQKV = (ROWS / 16) * (DIM / 16);   // 4096 tiles
  gemm_wmma<EPI_F16><<<tilesQKV / 8, 256, 0, stream>>>(Xh, Wq_t, Qh, nullptr, ROWS, DIM, DIM);
  gemm_wmma<EPI_F16><<<tilesQKV / 8, 256, 0, stream>>>(Xh, Wk_t, Kh, nullptr, ROWS, DIM, DIM);
  gemm_wmma<EPI_F16><<<tilesQKV / 8, 256, 0, stream>>>(Xh, Wv_t, Vh, nullptr, ROWS, DIM, DIM);

  // 3) transpose V per batch -> Vt (DIM x SEQ) for contiguous B-fragments
  dim3 tg(DIM / 32, SEQ / 32, BATCH), tb(32, 8);
  transpose_h<<<tg, tb, 0, stream>>>(Vh, Vt, SEQ, DIM);

  // 4) workgroup-cooperative flash attention (TDM-fed, dominant 34 GFLOP)
  dim3 ag(SEQ / 128, HEADS, BATCH);
  attn_wmma<<<ag, 256, 0, stream>>>(Qh, Kh, Vt, AttnO);

  // 5) output projection + residual (K = 2048)
  const int tilesOut = (ROWS / 16) * (DIM / 16);
  gemm_wmma<EPI_RESID_F32><<<tilesOut / 8, 256, 0, stream>>>(AttnO, Wo_t, Y0, x,
                                                             ROWS, DIM, HEADS * DIM);
  // 6) LN1 -> f16 for FFN
  layernorm_k<<<ROWS, DIM, 0, stream>>>(Y0, gamma1, nullptr, Y1h);

  // 7) FFN up + ReLU, FFN down
  const int tilesF1 = (ROWS / 16) * (FF / 16);
  gemm_wmma<EPI_RELU_F16><<<tilesF1 / 8, 256, 0, stream>>>(Y1h, Wf1_t, Hb, nullptr, ROWS, FF, DIM);
  gemm_wmma<EPI_F32><<<tilesOut / 8, 256, 0, stream>>>(Hb, Wf2_t, Y2, nullptr, ROWS, DIM, FF);

  // 8) final LN -> f32 output
  layernorm_k<<<ROWS, DIM, 0, stream>>>(Y2, gamma2, (float*)d_out, nullptr);
}
